// HistogramLoss_76957224010108
// MI455X (gfx1250) — matrix operand order repaired and checked by reading, not compiled, not run on hardware
//
#include <hip/hip_runtime.h>
#include <hip/hip_bf16.h>

// HistogramLoss on gfx1250 (MI455X):
//   dists = F F^T via v_wmma_f32_16x16x32_bf16, 128x64 block tiles,
//   upper-triangular tiles only (skip when bj < 2*bi).
//   Soft-binned pos/neg histograms via LDS ds_add_f32 -> global f32 atomics.
//   Finalize: normalize, 150-bin CDF, overlap dot -> scalar loss.

#define N_PTS   4096
#define DIMS    512
#define NSTEPS  150
#define TILE_M  128     // rows (I) per block: 8 waves x 16-row strip
#define TILE_N  64      // cols (J) per block: 4 x 16-col WMMA subtiles per wave
#define KSTEP   32
#define LDS_STRIDE 40   // 32 + 8 halves pad to spread LDS banks (80B rows, 16B aligned)

typedef __bf16 bf16_t;
typedef __attribute__((ext_vector_type(8)))  __bf16 v8bf;
typedef __attribute__((ext_vector_type(16))) __bf16 v16bf;
typedef __attribute__((ext_vector_type(8)))  float  v8f;

__device__ __forceinline__ v16bf load_frag16(const bf16_t* p0, const bf16_t* p1) {
    v8bf a = *(const v8bf*)p0;   // ds_load_b128
    v8bf b = *(const v8bf*)p1;   // ds_load_b128
    return __builtin_shufflevector(a, b, 0,1,2,3,4,5,6,7,8,9,10,11,12,13,14,15);
}

__device__ __forceinline__ v8bf cvt8(float4 a, float4 b) {
    v8bf r;
    r[0]=(__bf16)a.x; r[1]=(__bf16)a.y; r[2]=(__bf16)a.z; r[3]=(__bf16)a.w;
    r[4]=(__bf16)b.x; r[5]=(__bf16)b.y; r[6]=(__bf16)b.z; r[7]=(__bf16)b.w;
    return r;
}

__global__ void zero_acc_kernel(float* gAcc) {
    int t = blockIdx.x * blockDim.x + threadIdx.x;
    if (t < 512) gAcc[t] = 0.0f;
}

__global__ __launch_bounds__(256)
void hist_tiles_kernel(const float* __restrict__ feats,
                       const int*   __restrict__ classes,
                       float*       __restrict__ gAcc) {
    const int bi = blockIdx.y;                 // tile row (I), 128-row tiles
    const int bj = blockIdx.x;                 // tile col (J), 64-col tiles
    if (bj < 2 * bi) return;                   // tile entirely below diagonal: no i<j pairs
    const int I0 = bi * TILE_M, J0 = bj * TILE_N;

    __shared__ bf16_t ldsA[TILE_M * LDS_STRIDE];   // F[I0..I0+127, k0..k0+31]
    __shared__ bf16_t ldsB[TILE_N * LDS_STRIDE];   // F[J0..J0+63,  k0..k0+31]
    __shared__ float  sHist[308];  // [0..149]=pos, [152..301]=neg, [304]=cntP, [305]=cntN

    const int tid  = threadIdx.x;
    const int lane = tid & 31;
    const int wave = tid >> 5;                 // 0..7: 16-row strip of the I tile
    const int half = lane >> 4;                // 0: lanes 0-15, 1: lanes 16-31
    const int mr   = lane & 15;

    for (int i = tid; i < 308; i += 256) sHist[i] = 0.0f;

    // Staging: A tile 128x32 -> thread t loads 16 f32 (64B) of one row;
    //          B tile  64x32 -> thread t loads  8 f32 (32B) of one row.
    const int srowA = tid >> 1;                // 0..127
    const int scolA = (tid & 1) * 16;          // 0 or 16
    const int srowB = tid >> 2;                // 0..63
    const int scolB = (tid & 3) * 8;           // 0,8,16,24
    const float* gA = feats + (size_t)(I0 + srowA) * DIMS + scolA;
    const float* gB = feats + (size_t)(J0 + srowB) * DIMS + scolB;
    bf16_t* sA = &ldsA[srowA * LDS_STRIDE + scolA];
    bf16_t* sB = &ldsB[srowB * LDS_STRIDE + scolB];

    v8f acc[4] = {{}, {}, {}, {}};             // four 16x16 subtiles along N

    for (int k0 = 0; k0 < DIMS; k0 += KSTEP) {
        __syncthreads();                       // previous compute done before LDS overwrite
        float4 fa0 = *(const float4*)(gA + k0);
        float4 fa1 = *(const float4*)(gA + k0 + 4);
        float4 fa2 = *(const float4*)(gA + k0 + 8);
        float4 fa3 = *(const float4*)(gA + k0 + 12);
        float4 fb0 = *(const float4*)(gB + k0);
        float4 fb1 = *(const float4*)(gB + k0 + 4);
        if (k0 + KSTEP < DIMS) {               // global_prefetch_b8 for next K panel
            __builtin_prefetch(gA + k0 + KSTEP, 0, 1);
            __builtin_prefetch(gB + k0 + KSTEP, 0, 1);
        }
        *(v8bf*)sA       = cvt8(fa0, fa1);
        *(v8bf*)(sA + 8) = cvt8(fa2, fa3);
        *(v8bf*)sB       = cvt8(fb0, fb1);
        __syncthreads();

        // A-frag (16x32): lanes 0-15 hold K 0-7 & 16-23; lanes 16-31 hold K 8-15 & 24-31
        const bf16_t* arow = &ldsA[(wave * 16 + mr) * LDS_STRIDE];
        v16bf afrag = load_frag16(arow + half * 8, arow + half * 8 + 16);

        // B-frags (32x16): B[k][n] = F_J[n][k]; lane's column n = row J0+n of F,
        // lanes 0-15 hold K 0-7 & 16-23; lanes 16-31 hold K 8-15 & 24-31 (matches A)
#pragma unroll
        for (int sub = 0; sub < 4; ++sub) {
            const bf16_t* brow = &ldsB[(sub * 16 + mr) * LDS_STRIDE + half * 8];
            v16bf bfrag = load_frag16(brow, brow + 16);
            acc[sub] = __builtin_amdgcn_wmma_f32_16x16x32_bf16(
                false, afrag, false, bfrag, (short)0, acc[sub], false, false);
        }
    }

    // Epilogue: C/D layout -> VGPR r holds (M = half*8 + r, N = mr)
    const int i_base = I0 + wave * 16 + half * 8;
    int ci[8];
#pragma unroll
    for (int r = 0; r < 8; ++r) ci[r] = classes[i_base + r];

    const float inv_step = (NSTEPS - 1) * 0.5f;   // 1/step = 149/2
    float cP = 0.0f, cN = 0.0f;
#pragma unroll
    for (int sub = 0; sub < 4; ++sub) {
        const int j  = J0 + sub * 16 + mr;
        const int cj = classes[j];
#pragma unroll
        for (int r = 0; r < 8; ++r) {
            const int i = i_base + r;
            if (i < j) {
                float s = acc[sub][r];
                float t = (s + 1.0f) * inv_step;
                int idx = (int)floorf(t);
                idx = min(max(idx, 0), NSTEPS - 1);
                float frac = t - (float)idx;          // == (s - (idx*step-1))/step
                int idx1 = min(idx + 1, NSTEPS - 1);
                bool pos = (ci[r] == cj);
                float* h = pos ? sHist : (sHist + 152);
                atomicAdd(&h[idx],  1.0f - frac);     // ds_add_f32
                atomicAdd(&h[idx1], frac);
                if (pos) cP += 1.0f; else cN += 1.0f;
            }
        }
    }
    if (cP != 0.0f) atomicAdd(&sHist[304], cP);
    if (cN != 0.0f) atomicAdd(&sHist[305], cN);
    __syncthreads();

    for (int b = tid; b < NSTEPS; b += 256) {
        atomicAdd(&gAcc[b],          sHist[b]);       // global_atomic_add_f32
        atomicAdd(&gAcc[NSTEPS + b], sHist[152 + b]);
    }
    if (tid == 0) {
        atomicAdd(&gAcc[300], sHist[304]);
        atomicAdd(&gAcc[301], sHist[305]);
    }
}

__global__ void finalize_kernel(const float* __restrict__ gAcc, float* __restrict__ out) {
    if (threadIdx.x == 0 && blockIdx.x == 0) {
        const float sp = gAcc[300];
        const float sn = gAcc[301];
        float cdf = 0.0f, loss = 0.0f;
        for (int b = 0; b < NSTEPS; ++b) {
            cdf  += gAcc[b] / sp;                  // cumsum of normalized pos hist
            loss += (gAcc[NSTEPS + b] / sn) * cdf; // overlap with normalized neg hist
        }
        out[0] = loss;
    }
}

extern "C" void kernel_launch(void* const* d_in, const int* in_sizes, int n_in,
                              void* d_out, int out_size, void* d_ws, size_t ws_size,
                              hipStream_t stream) {
    const float* feats   = (const float*)d_in[0];
    const int*   classes = (const int*)d_in[1];
    float* gAcc = (float*)d_ws;   // [0..149] pos, [150..299] neg, [300] cntP, [301] cntN
    float* out  = (float*)d_out;

    zero_acc_kernel<<<1, 512, 0, stream>>>(gAcc);
    dim3 grid(N_PTS / TILE_N, N_PTS / TILE_M);   // (64, 32); lower tiles early-exit
    hist_tiles_kernel<<<grid, 256, 0, stream>>>(feats, classes, gAcc);
    finalize_kernel<<<1, 32, 0, stream>>>(gAcc, out);
}
